// GRUD_58566174048663
// MI455X (gfx1250) — matrix operand order repaired and checked by reading, not compiled, not run on hardware
//
#include <hip/hip_runtime.h>
#include <hip/hip_bf16.h>
#include <math.h>
#include <stdint.h>

// ---------------------------------------------------------------------------
// GRU-D persistent recurrence for MI455X (gfx1250, wave32, WMMA bf16->f32)
//   B=512, T=200, D=128, H=512, COMB=768
//   Grid: 8 batch-groups x 16 H-slice WGs = 128 blocks, 256 threads (8 waves)
//   - Gate weights live in LDS as bf16 for all 200 steps.
//   - Per-step hidden-state exchange goes through L2 in bf16 and is pulled
//     back into the LDS comb buffer by the Tensor Data Mover (TDM) with LDS
//     padding to hit the 1536-B comb row stride (tensor_load_to_lds +
//     s_wait_tensorcnt), instead of per-thread scalar loads/converts.
// ---------------------------------------------------------------------------

typedef __bf16 bf16;
typedef __attribute__((ext_vector_type(16))) __bf16    v16bf;
typedef __attribute__((ext_vector_type(8)))  __bf16    v8bf;
typedef __attribute__((ext_vector_type(8)))  float     v8f;
typedef __attribute__((ext_vector_type(4)))  unsigned  v4u;
typedef __attribute__((ext_vector_type(8)))  unsigned  v8u;

#define B_     512
#define T_     200
#define D_     128
#define H_     512
#define COMB_  768
#define BTILE  64          // batch rows per group
#define HSLICE 32          // hidden cols per workgroup
#define WPG    16          // workgroups per batch group (H_/HSLICE)
#define NGRP   8           // batch groups (B_/BTILE)
#define NBLK   (WPG*NGRP)  // 128
#define NTHR   256         // 8 wave32 per block
#define WSLICE (HSLICE*COMB_)  // 24576 elems per gate slice

// dynamic LDS layout (bytes)
constexpr size_t SMEM_BYTES =
    (size_t)(3*WSLICE + HSLICE*D_ + BTILE*COMB_) * sizeof(bf16)   // 253952
  + (size_t)(BTILE*HSLICE + 4*HSLICE + 2*D_) * sizeof(float);     // + 9728

// cross-workgroup exchange buffers (fully rewritten every launch before read)
__device__ bf16     g_hx[B_*H_];     // decayed h, bf16 (TDM-copied into LDS)
__device__ bf16     g_rx[B_*H_];     // r * h, bf16    (TDM-copied into LDS)
__device__ float    g_hfin[B_*H_];   // final h (f32) for the FC head
__device__ unsigned g_bar[NGRP];     // per-group arrival counters

// ---------------------------------------------------------------------------

__device__ __forceinline__ v8f wmma_bf16(v16bf a, v16bf b, v8f c) {
  // D = A(16x32) * B(32x16) + C(16x16 f32)
  return __builtin_amdgcn_wmma_f32_16x16x32_bf16(
      /*neg_a=*/false, a, /*neg_b=*/false, b,
      /*c_mod=*/(short)0, c, /*reuse_a=*/false, /*reuse_b=*/false);
}

__device__ __forceinline__ v16bf frag16(const bf16* p0, const bf16* p1) {
  v8bf lo = *(const v8bf*)p0;
  v8bf hi = *(const v8bf*)p1;
  v16bf r;
#pragma unroll
  for (int i = 0; i < 8; ++i) { r[i] = lo[i]; r[8 + i] = hi[i]; }
  return r;
}

// A fragment (16x32 bf16): lane half h: elems 0..15 -> K=16h..16h+15 contiguous
// (same k order as B).
__device__ __forceinline__ v16bf loadA(const bf16* rowp, int half) {
  return frag16(rowp + 16*half, rowp + 16*half + 8);
}
// B fragment (32x16 bf16): lane half h: elems 0..15 -> K=16h..16h+15 contiguous.
__device__ __forceinline__ v16bf loadB(const bf16* p) {
  return frag16(p, p + 8);
}

__device__ __forceinline__ void group_barrier(int grp) {
  __threadfence();          // make this thread's global stores visible
  __syncthreads();          // all waves' stores issued + fenced
  if (threadIdx.x == 0) {
    unsigned t      = atomicAdd(&g_bar[grp], 1u);           // device scope
    unsigned target = t - (t % WPG) + WPG;                  // end of this round
    while (__hip_atomic_load(&g_bar[grp], __ATOMIC_ACQUIRE,
                             __HIP_MEMORY_SCOPE_AGENT) < target) {
      __builtin_amdgcn_s_sleep(1);
    }
  }
  __syncthreads();
}

// Issue a TDM 2D tile load: 64 rows x 1024B from `gaddr` (row stride 1024B in
// global) into LDS at `ldsAddr`, inserting 512B of LDS padding after each
// 1024B row so rows land at the 1536B comb stride. Wave0-only, then
// s_wait_tensorcnt==0 + __syncthreads() releases the block.
__device__ __forceinline__ void tdm_load_tile(unsigned ldsAddr,
                                              unsigned long long gaddr,
                                              bool waveZero) {
  if (waveZero) {
    // D# group 0: count=1, lds_addr, global_addr[56:0], type=2
    v4u g0;
    g0[0] = 1u;                                    // count = 1 (valid)
    g0[1] = ldsAddr;                               // LDS byte address
    g0[2] = (unsigned)gaddr;                       // global addr [31:0]
    g0[3] = ((unsigned)(gaddr >> 32) & 0x01FFFFFFu) | (2u << 30); // [56:32]|type=2
    // D# group 1:
    //  data_size=2B (1<<16), pad_enable (1<<20), pad_interval=7 (256 DW),
    //  pad_amount=127 (128 DW = 512B)
    v8u g1;
    g1[0] = (1u << 16) | (1u << 20) | (7u << 22) | (127u << 25); // 0xFFD10000
    g1[1] = (unsigned)(H_) << 16;      // tensor_dim0 = 512 (low16 into [63:48])
    g1[2] = (unsigned)(BTILE) << 16;   // tensor_dim0 hi=0 | tensor_dim1 = 64
    g1[3] = (unsigned)(H_) << 16;      // tensor_dim1 hi=0 | tile_dim0 = 512
    g1[4] = (unsigned)(BTILE);         // tile_dim1 = 64 | tile_dim2 = 0
    g1[5] = (unsigned)(H_);            // tensor_dim0_stride = 512 (low 32)
    g1[6] = 0u;                        // stride0 hi | tensor_dim1_stride lo
    g1[7] = 0u;                        // tensor_dim1_stride hi
    asm volatile("tensor_load_to_lds %0, %1"
                 :: "s"(g0), "s"(g1)
                 : "memory");
    __builtin_amdgcn_s_wait_tensorcnt(0);
  }
  __syncthreads();
}

__global__ void grud_reset() {
  if (threadIdx.x < NGRP) g_bar[threadIdx.x] = 0u;
}

// ---------------------------------------------------------------------------

__global__ void __launch_bounds__(NTHR, 1)
grud_persistent(const float* __restrict__ X,  const float* __restrict__ XL,
                const float* __restrict__ Mk, const float* __restrict__ Dl,
                const float* __restrict__ Xm,
                const float* __restrict__ wgx, const float* __restrict__ bgx,
                const float* __restrict__ Wgh, const float* __restrict__ bgh,
                const float* __restrict__ Wz,  const float* __restrict__ bz,
                const float* __restrict__ Wr,  const float* __restrict__ br,
                const float* __restrict__ Wh,  const float* __restrict__ bh)
{
  extern __shared__ char smem[];
  bf16*  sW    = (bf16*)smem;                   // [3][HSLICE][COMB] gate weights
  bf16*  sWgh  = sW + 3*WSLICE;                 // [HSLICE][D]
  bf16*  sComb = sWgh + HSLICE*D_;              // [BTILE][COMB] = [x | h | m]
  float* sH    = (float*)(sComb + BTILE*COMB_); // [BTILE][HSLICE] h slice (f32)
  float* sBz   = sH + BTILE*HSLICE;             // [HSLICE]
  float* sBr   = sBz + HSLICE;
  float* sBh   = sBr + HSLICE;
  float* sBgh  = sBh + HSLICE;
  float* sWgx  = sBgh + HSLICE;                 // [D]
  float* sBgx  = sWgx + D_;                     // [D]

  const int tid   = threadIdx.x;
  const int grp   = blockIdx.x >> 4;    // batch group
  const int wgid  = blockIdx.x & 15;    // H-slice within group
  const int gbase = grp  * BTILE;       // first batch row
  const int hbase = wgid * HSLICE;      // first hidden column

  // LDS byte address of the comb "middle" region (column 128 of row 0),
  // relative to the workgroup LDS base (static LDS + dynamic offset).
  const unsigned combMidLds =
      __builtin_amdgcn_groupstaticsize()
    + (unsigned)((3*WSLICE + HSLICE*D_) * sizeof(bf16))   // sComb offset
    + (unsigned)(D_ * sizeof(bf16));                      // +256B (x region)
  const unsigned long long gaH  = (unsigned long long)(uintptr_t)(g_hx + (size_t)gbase*H_);
  const unsigned long long gaRH = (unsigned long long)(uintptr_t)(g_rx + (size_t)gbase*H_);

  // ---- prologue: stage weight slices to LDS as bf16, zero h ----
  for (int i = tid; i < HSLICE*COMB_; i += NTHR) {
    int r = i / COMB_, c = i - r*COMB_;
    size_t gi = (size_t)(hbase + r)*COMB_ + c;
    sW[i]            = (bf16)Wz[gi];
    sW[WSLICE + i]   = (bf16)Wr[gi];
    sW[2*WSLICE + i] = (bf16)Wh[gi];
  }
  for (int i = tid; i < HSLICE*D_; i += NTHR) {
    int r = i >> 7, c = i & (D_-1);
    sWgh[i] = (bf16)Wgh[(size_t)(hbase + r)*D_ + c];
  }
  for (int i = tid; i < HSLICE; i += NTHR) {
    sBz[i]  = bz[hbase + i];  sBr[i]  = br[hbase + i];
    sBh[i]  = bh[hbase + i];  sBgh[i] = bgh[hbase + i];
  }
  for (int i = tid; i < D_; i += NTHR) { sWgx[i] = wgx[i]; sBgx[i] = bgx[i]; }
  for (int i = tid; i < BTILE*HSLICE; i += NTHR) sH[i] = 0.0f;
  __syncthreads();

  // ---- per-wave WMMA tile assignment: 4 M-tiles x 2 N-tiles ----
  const int wave  = tid >> 5, lane = tid & 31;
  const bool w0   = (wave == 0);
  const int rbase = (wave >> 1) * 16;           // tile row base in [0,64)
  const int cbase = (wave & 1)  * 16;           // tile col base in [0,32)
  const int lm    = lane & 15;                  // A row / B-D column lane
  const int half  = lane >> 4;
  const int arow  = rbase + lm;                 // this lane's A row
  const int cl    = cbase + lm;                 // this lane's output column

  const bf16* aRow = sComb + arow*COMB_;
  const bf16* bzP  = sW + 0*WSLICE + cl*COMB_ + 16*half;
  const bf16* brP  = sW + 1*WSLICE + cl*COMB_ + 16*half;
  const bf16* bhP  = sW + 2*WSLICE + cl*COMB_ + 16*half;
  const bf16* bgP  = sWgh          + cl*D_    + 16*half;

#pragma unroll 1
  for (int t = 0; t < T_; ++t) {
    __syncthreads();   // previous step's readers of sComb are done

    // (a) elementwise: input decay + imputation; stage x/d/m into comb (bf16)
    for (int i = tid; i < BTILE*D_; i += NTHR) {
      int row = i >> 7, k = i & (D_-1);
      size_t gi = ((size_t)(gbase + row)*T_ + t)*D_ + k;
      float d  = Dl[gi];
      float x  = X[gi];
      float xl = XL[gi];
      float m  = Mk[gi];
      float xm = Xm[(size_t)t*D_ + k];
      float dx = __expf(-fmaxf(d*sWgx[k] + sBgx[k], 0.0f));
      float xi = m*x + (1.0f - m)*(dx*xl + (1.0f - dx)*xm);
      bf16* crow = sComb + row*COMB_;
      crow[k]       = (bf16)xi;    // comb[:, 0:128)   = x
      crow[D_ + k]  = (bf16)d;     // temp staging of delta for gamma_h GEMM
      crow[640 + k] = (bf16)m;     // comb[:, 640:768) = mask
      if ((t + 1 < T_) && ((k & 15) == 0)) {   // prefetch next step's rows
        size_t gn = gi + D_;
        __builtin_prefetch(&Dl[gn], 0, 1);
        __builtin_prefetch(&X[gn],  0, 1);
        __builtin_prefetch(&XL[gn], 0, 1);
        __builtin_prefetch(&Mk[gn], 0, 1);
      }
    }
    __syncthreads();

    // (b) gamma_h GEMM: delta_h = exp(-relu(d @ Wgh^T + bgh)), K = 128
    v8f accG = {0,0,0,0,0,0,0,0};
    {
      const bf16* aD = aRow + D_;   // delta staged at comb cols [128,256)
#pragma unroll
      for (int kk = 0; kk < D_; kk += 32)
        accG = wmma_bf16(loadA(aD + kk, half), loadB(bgP + kk), accG);
    }
    // (c) decay hidden state slice, publish (bf16) to group
#pragma unroll
    for (int r = 0; r < 8; ++r) {
      int rl = rbase + 8*half + r;                       // D-matrix M = r + 8*half
      float dh = __expf(-fmaxf(accG[r] + sBgh[cl], 0.0f));
      float hd = dh * sH[rl*HSLICE + cl];
      sH[rl*HSLICE + cl] = hd;
      g_hx[(size_t)(gbase + rl)*H_ + hbase + cl] = (bf16)hd;
    }

    group_barrier(grp);   // barrier 1: full decayed h available in g_hx

    // (e) TDM: comb[:,128:640) <- decayed h tile (bf16, padded rows)
    tdm_load_tile(combMidLds, gaH, w0);

    // (f) z/r GEMMs over K = 768 (shared A fragment per k-step)
    v8f accZ = {0,0,0,0,0,0,0,0};
    v8f accR = {0,0,0,0,0,0,0,0};
#pragma unroll 4
    for (int kk = 0; kk < COMB_; kk += 32) {
      v16bf a = loadA(aRow + kk, half);
      accZ = wmma_bf16(a, loadB(bzP + kk), accZ);
      accR = wmma_bf16(a, loadB(brP + kk), accR);
    }
    // (g) gates; publish r*h (bf16) for comb2
    float zv[8];
#pragma unroll
    for (int r = 0; r < 8; ++r) {
      int rl = rbase + 8*half + r;
      zv[r]    = 1.0f / (1.0f + __expf(-(accZ[r] + sBz[cl])));
      float rv = 1.0f / (1.0f + __expf(-(accR[r] + sBr[cl])));
      g_rx[(size_t)(gbase + rl)*H_ + hbase + cl] = (bf16)(rv * sH[rl*HSLICE + cl]);
    }

    group_barrier(grp);   // barrier 2: full r*h available in g_rx

    // (i) TDM: comb[:,128:640) <- r*h tile
    tdm_load_tile(combMidLds, gaRH, w0);

    // (j) h_tilde GEMM over K = 768
    v8f accH = {0,0,0,0,0,0,0,0};
#pragma unroll 4
    for (int kk = 0; kk < COMB_; kk += 32)
      accH = wmma_bf16(loadA(aRow + kk, half), loadB(bhP + kk), accH);

    // (k) hidden update (own slice only)
#pragma unroll
    for (int r = 0; r < 8; ++r) {
      int rl = rbase + 8*half + r;
      float ht = tanhf(accH[r] + sBh[cl]);
      float hd = sH[rl*HSLICE + cl];
      float hn = (1.0f - zv[r])*hd + zv[r]*ht;
      sH[rl*HSLICE + cl] = hn;
      if (t == T_ - 1)
        g_hfin[(size_t)(gbase + rl)*H_ + hbase + cl] = hn;  // final h (f32)
    }
  }
}

// ---------------------------------------------------------------------------
// Final FC [512,512]x[512,2] + train-mode BatchNorm over the batch dim.
// ---------------------------------------------------------------------------
__global__ void grud_fcbn(const float* __restrict__ Wfc, const float* __restrict__ bfc,
                          const float* __restrict__ gam, const float* __restrict__ bet,
                          float* __restrict__ out)
{
  __shared__ float red[B_];
  const int b = threadIdx.x;                 // 512 threads == batch
  const float* h = g_hfin + (size_t)b*H_;
  float l0 = bfc[0], l1 = bfc[1];
  for (int k = 0; k < H_; ++k) {
    float hv = h[k];
    l0 += hv * Wfc[k];
    l1 += hv * Wfc[H_ + k];
  }
  float lg[2] = {l0, l1};
#pragma unroll 1
  for (int c = 0; c < 2; ++c) {
    red[b] = lg[c]; __syncthreads();
    for (int s = B_/2; s > 0; s >>= 1) { if (b < s) red[b] += red[b + s]; __syncthreads(); }
    float mu = red[0] * (1.0f / B_); __syncthreads();
    float dlt = lg[c] - mu;
    red[b] = dlt * dlt; __syncthreads();
    for (int s = B_/2; s > 0; s >>= 1) { if (b < s) red[b] += red[b + s]; __syncthreads(); }
    float var = red[0] * (1.0f / B_); __syncthreads();
    out[b*2 + c] = gam[c] * dlt * rsqrtf(var + 1e-5f) + bet[c];
  }
}

// ---------------------------------------------------------------------------

extern "C" void kernel_launch(void* const* d_in, const int* in_sizes, int n_in,
                              void* d_out, int out_size, void* d_ws, size_t ws_size,
                              hipStream_t stream) {
  (void)in_sizes; (void)n_in; (void)d_ws; (void)ws_size; (void)out_size;
  const float* X   = (const float*)d_in[0];
  const float* XL  = (const float*)d_in[1];
  const float* Mk  = (const float*)d_in[2];
  const float* Dl  = (const float*)d_in[3];
  const float* Xm  = (const float*)d_in[4];
  const float* wgx = (const float*)d_in[5];
  const float* bgx = (const float*)d_in[6];
  const float* Wgh = (const float*)d_in[7];
  const float* bgh = (const float*)d_in[8];
  const float* Wz  = (const float*)d_in[9];
  const float* bz  = (const float*)d_in[10];
  const float* Wr  = (const float*)d_in[11];
  const float* br  = (const float*)d_in[12];
  const float* Wh  = (const float*)d_in[13];
  const float* bh  = (const float*)d_in[14];
  const float* Wfc = (const float*)d_in[15];
  const float* bfc = (const float*)d_in[16];
  const float* gam = (const float*)d_in[17];
  const float* bet = (const float*)d_in[18];

  // Allow >64KB dynamic LDS (WGP has 320KB). Ignore errors (idempotent).
  (void)hipFuncSetAttribute((const void*)grud_persistent,
                            hipFuncAttributeMaxDynamicSharedMemorySize,
                            (int)SMEM_BYTES);

  grud_reset<<<dim3(1), dim3(32), 0, stream>>>();
  grud_persistent<<<dim3(NBLK), dim3(NTHR), SMEM_BYTES, stream>>>(
      X, XL, Mk, Dl, Xm, wgx, bgx, Wgh, bgh, Wz, bz, Wr, br, Wh, bh);
  grud_fcbn<<<dim3(1), dim3(B_), 0, stream>>>(Wfc, bfc, gam, bet, (float*)d_out);
}
